// GraphUpdateBlock_89412629168730
// MI455X (gfx1250) — matrix operand order repaired and checked by reading, not compiled, not run on hardware
//
#include <hip/hip_runtime.h>

typedef __attribute__((ext_vector_type(16))) __bf16 v16bf;
typedef __attribute__((ext_vector_type(8)))  float  v8f;

#define NODES 8192
#define E_IN  131072
#define E_EXT (E_IN + NODES)   // 139264
#define IN1   770
#define IN1P  800              // padded to multiple of 32 for guard-free GEMM

// ---------- helpers ----------
__device__ __forceinline__ unsigned int bfbits(float f) {
  unsigned int u = __float_as_uint(f);
  return (u + 0x7FFFu + ((u >> 16) & 1u)) >> 16;   // RNE
}
__device__ __forceinline__ unsigned int packbf(float lo, float hi) {
  return bfbits(lo) | (bfbits(hi) << 16);
}
__device__ __forceinline__ __bf16 f2bf(float f) {
  unsigned short s = (unsigned short)bfbits(f);
  __bf16 r; __builtin_memcpy(&r, &s, sizeof(r));
  return r;
}

struct BF16x16 { uint4 a, b; };   // 32 bytes == v16bf

__device__ __forceinline__ v16bf ldfrag_a(const __bf16* rp, int kh) {
  // A row chunk: K = kh*8..kh*8+7 and 16+kh*8..16+kh*8+7 (ISA 7.12.2)
  BF16x16 t;
  t.a = *reinterpret_cast<const uint4*>(rp + kh * 8);
  t.b = *reinterpret_cast<const uint4*>(rp + 16 + kh * 8);
  return __builtin_bit_cast(v16bf, t);
}
__device__ __forceinline__ v16bf ldfrag_b(const __bf16* cp, int kh) {
  // B column chunk: K = kh*8..kh*8+7 and 16+kh*8..16+kh*8+7 (matches A)
  BF16x16 t;
  t.a = *reinterpret_cast<const uint4*>(cp + kh * 8);
  t.b = *reinterpret_cast<const uint4*>(cp + 16 + kh * 8);
  return __builtin_bit_cast(v16bf, t);
}

__device__ __forceinline__ void atomicMaxF(float* addr, float v) {
  if (v >= 0.0f) atomicMax((int*)addr, __float_as_int(v));
  else           atomicMin((unsigned int*)addr, __float_as_uint(v));
}
__device__ __forceinline__ float sigmoidf(float x) { return 1.0f / (1.0f + expf(-x)); }
__device__ __forceinline__ float softplusf(float x) {
  return fmaxf(x, 0.0f) + log1pf(expf(-fabsf(x)));
}
__device__ __forceinline__ float waveRedSum(float v) {
  for (int off = 16; off > 0; off >>= 1) v += __shfl_down(v, off, 32);
  return v;
}

// ---------- WMMA GEMM: C = act(A[MxK] @ W[KxN] + bias) ----------
// Requires M%32==0, N%128==0, K%32==0. Block = 256 threads (8 waves).
// Block tile 32(M)x128(N); wave (wm,wn) = (w>>2, w&3) computes rows wm*16..+16,
// cols wn*32..+32 as two 16x16 WMMA tiles sharing one A fragment.
template <int ACT>  // 0 = none, 1 = relu
__global__ __launch_bounds__(256) void gemm_bf16_wmma(
    const float* __restrict__ A, const float* __restrict__ W,
    const float* __restrict__ bias, float* __restrict__ C,
    int M, int N, int K) {
  __shared__ alignas(16) __bf16 As[32 * 48];    // [row][k], stride 48
  __shared__ alignas(16) __bf16 Bt[128 * 48];   // [col][k], stride 48 (transposed)

  const int tid  = threadIdx.x;
  const int lane = tid & 31;
  const int wave = tid >> 5;
  const int wm = wave >> 2, wn = wave & 3;
  const int rowBase = blockIdx.y * 32;
  const int colBase = blockIdx.x * 128;

  v8f acc0 = {}, acc1 = {};

  for (int k0 = 0; k0 < K; k0 += 32) {
    // A tile 32x32: one float4 per thread, packed u32x2 store
    {
      const int r = tid >> 3, c4 = tid & 7;
      const float4 v = *reinterpret_cast<const float4*>(
          A + (size_t)(rowBase + r) * K + k0 + c4 * 4);
      *reinterpret_cast<uint2*>(&As[r * 48 + c4 * 4]) =
          make_uint2(packbf(v.x, v.y), packbf(v.z, v.w));
    }
    // B tile 32x128: 4 float4 per thread (coalesced), scatter into transposed LDS
#pragma unroll
    for (int j = 0; j < 4; ++j) {
      const int u = tid + j * 256;
      const int r = u >> 5, c4 = u & 31;
      const float4 v = *reinterpret_cast<const float4*>(
          W + (size_t)(k0 + r) * N + colBase + c4 * 4);
      const int cb = c4 * 4;
      Bt[(cb + 0) * 48 + r] = f2bf(v.x);
      Bt[(cb + 1) * 48 + r] = f2bf(v.y);
      Bt[(cb + 2) * 48 + r] = f2bf(v.z);
      Bt[(cb + 3) * 48 + r] = f2bf(v.w);
    }
    __syncthreads();

    const int m  = lane & 15;
    const int kh = lane >> 4;
    const v16bf afrag = ldfrag_a(&As[(wm * 16 + m) * 48], kh);
    const v16bf b0 = ldfrag_b(&Bt[(wn * 32 + m) * 48], kh);
    const v16bf b1 = ldfrag_b(&Bt[(wn * 32 + 16 + m) * 48], kh);
    acc0 = __builtin_amdgcn_wmma_f32_16x16x32_bf16(false, afrag, false, b0,
                                                   (short)0, acc0, false, false);
    acc1 = __builtin_amdgcn_wmma_f32_16x16x32_bf16(false, afrag, false, b1,
                                                   (short)0, acc1, false, false);
    __syncthreads();
  }

  // C layout: VGPR r -> row (lane<16 ? r : r+8), col = lane&15
  const int colb = colBase + wn * 32 + (lane & 15);
  const int rowb = rowBase + wm * 16 + (lane >> 4) * 8;
  const float bb0 = bias[colb], bb1 = bias[colb + 16];
#pragma unroll
  for (int r = 0; r < 8; ++r) {
    const int row = rowb + r;
    float v0 = acc0[r] + bb0;
    float v1 = acc1[r] + bb1;
    if (ACT == 1) { v0 = fmaxf(v0, 0.0f); v1 = fmaxf(v1, 0.0f); }
    C[(size_t)row * N + colb]      = v0;
    C[(size_t)row * N + colb + 16] = v1;
  }
}

// ---------- elementwise / graph kernels ----------
__global__ void fill_f32(float* p, float v, int n) {
  int i = blockIdx.x * blockDim.x + threadIdx.x;
  if (i < n) p[i] = v;
}

__global__ void pad_weight(const float* __restrict__ in, float* __restrict__ out,
                           int Kin, int N, int Kpad) {
  int idx = blockIdx.x * blockDim.x + threadIdx.x;
  if (idx >= Kpad * N) return;
  const int k = idx / N;
  out[idx] = (k < Kin) ? in[idx] : 0.0f;
}

__global__ void build_x(const float* __restrict__ ct, const float* __restrict__ cs,
                        const float* __restrict__ ep, const float* __restrict__ fl,
                        float* __restrict__ X) {
  int idx = blockIdx.x * blockDim.x + threadIdx.x;
  if (idx >= NODES * IN1P) return;
  const int i = idx / IN1P, j = idx - i * IN1P;
  float v;
  if (j < 256)      v = ct[i * 256 + j];
  else if (j < 512) v = cs[i * 256 + (j - 256)];
  else if (j < 514) v = ep[i * 2 + (j - 512)];
  else if (j < 770) v = fl[i * 256 + (j - 514)];
  else              v = 0.0f;   // K padding
  X[idx] = v;
}

__global__ void ea_mean(const float* __restrict__ ea, float* __restrict__ mean) {
  __shared__ float red[256];
  float s = 0.0f;
  for (int i = threadIdx.x; i < E_IN; i += 256) s += ea[i];
  red[threadIdx.x] = s;
  __syncthreads();
  for (int st = 128; st > 0; st >>= 1) {
    if (threadIdx.x < st) red[threadIdx.x] += red[threadIdx.x + st];
    __syncthreads();
  }
  if (threadIdx.x == 0) mean[0] = red[0] / (float)E_IN;
}

__global__ void build_edges(const long long* __restrict__ edges,
                            const float* __restrict__ eattr,
                            const float* __restrict__ mean,
                            int* __restrict__ src, int* __restrict__ dst,
                            float* __restrict__ ea) {
  int e = blockIdx.x * blockDim.x + threadIdx.x;
  if (e >= E_EXT) return;
  if (e < E_IN) {
    src[e] = (int)edges[e];
    dst[e] = (int)edges[E_IN + e];
    ea[e]  = eattr[e];
  } else {
    const int i = e - E_IN;     // self-loop
    src[e] = i; dst[e] = i; ea[e] = mean[0];
  }
}

// wave-per-(edge,head): coalesced reads + shfl reduction, then atomic segmax
__global__ void gat_score_w(const float* __restrict__ xl, const float* __restrict__ xr,
                            const float* __restrict__ ea, const int* __restrict__ src,
                            const int* __restrict__ dst, const float* __restrict__ We,
                            const float* __restrict__ att, float* __restrict__ esc,
                            float* __restrict__ segmax, int H, int C) {
  const int widx = (blockIdx.x * blockDim.x + threadIdx.x) >> 5;
  const int lane = threadIdx.x & 31;
  if (widx >= E_EXT * H) return;
  const int e = widx / H, h = widx - e * H;
  const int s = src[e], d = dst[e];
  const float eav = ea[e];
  const float* pl = xl + (size_t)s * H * C + h * C;
  const float* pr = xr + (size_t)d * H * C + h * C;
  const float* pw = We + h * C;
  const float* pa = att + h * C;
  float acc = 0.0f;
  for (int c = lane; c < C; c += 32) {
    float g = pl[c] + pr[c] + eav * pw[c];
    g = g > 0.0f ? g : 0.2f * g;
    acc += g * pa[c];
  }
  acc = waveRedSum(acc);
  if (lane == 0) {
    esc[widx] = acc;
    atomicMaxF(&segmax[d * H + h], acc);
  }
}

__global__ void gat_alpha(float* __restrict__ esc, const int* __restrict__ dst,
                          const float* __restrict__ segmax, float* __restrict__ segsum,
                          int H) {
  int idx = blockIdx.x * blockDim.x + threadIdx.x;
  if (idx >= E_EXT * H) return;
  const int e = idx / H, h = idx - e * H;
  const int d = dst[e];
  const float a = expf(esc[idx] - segmax[d * H + h]);
  esc[idx] = a;
  atomicAdd(&segsum[d * H + h], a);
}

// wave-per-(edge,head): coalesced gather of xl[src] + coalesced atomic scatter
__global__ void gat_agg_w(const float* __restrict__ aun, const int* __restrict__ src,
                          const int* __restrict__ dst, const float* __restrict__ segsum,
                          const float* __restrict__ xl, float* __restrict__ out,
                          int H, int C) {
  const int widx = (blockIdx.x * blockDim.x + threadIdx.x) >> 5;
  const int lane = threadIdx.x & 31;
  if (widx >= E_EXT * H) return;
  const int e = widx / H, h = widx - e * H;
  const int s = src[e], d = dst[e];
  const float alpha = aun[widx] / (segsum[d * H + h] + 1e-16f);
  const float* pl = xl + (size_t)s * H * C + h * C;
  float* po = out + (size_t)d * H * C + h * C;
  for (int c = lane; c < C; c += 32) atomicAdd(&po[c], alpha * pl[c]);
}

__global__ void add_bias_relu(float* __restrict__ x, const float* __restrict__ b,
                              int total, int N) {
  int idx = blockIdx.x * blockDim.x + threadIdx.x;
  if (idx >= total) return;
  float v = x[idx] + b[idx % N];
  x[idx] = v > 0.0f ? v : 0.0f;
}

// conv2 finalize: mean over H=4 heads + bias2
__global__ void head_mean(const float* __restrict__ out2, const float* __restrict__ bias,
                          float* __restrict__ x2) {
  int idx = blockIdx.x * blockDim.x + threadIdx.x;
  if (idx >= NODES * 256) return;
  const int i = idx >> 8, c = idx & 255;
  const float* p = out2 + (size_t)i * 1024 + c;
  x2[idx] = 0.25f * (p[0] + p[256] + p[512] + p[768]) + bias[c];
}

__global__ void gru_cell(const float* __restrict__ gi, const float* __restrict__ gh,
                         const float* __restrict__ hf, float* __restrict__ hnew) {
  int idx = blockIdx.x * blockDim.x + threadIdx.x;
  if (idx >= NODES * 256) return;
  const int i = idx >> 8, d = idx & 255;
  const float* pi = gi + (size_t)i * 768;
  const float* ph = gh + (size_t)i * 768;
  const float r = sigmoidf(pi[d] + ph[d]);
  const float z = sigmoidf(pi[256 + d] + ph[256 + d]);
  const float n = tanhf(pi[512 + d] + r * ph[512 + d]);
  hnew[idx] = (1.0f - z) * n + z * hf[idx];
}

// wave-per-row, two outputs (K=256)
template <int SIG>
__global__ void head2_w(const float* __restrict__ T, const float* __restrict__ W2,
                        const float* __restrict__ b2, float* __restrict__ out) {
  const int row = (blockIdx.x * blockDim.x + threadIdx.x) >> 5;
  const int lane = threadIdx.x & 31;
  if (row >= NODES) return;
  const float* t = T + (size_t)row * 256;
  float a0 = 0.0f, a1 = 0.0f;
  for (int k = lane; k < 256; k += 32) {
    const float tv = t[k];
    a0 += tv * W2[k * 2];
    a1 += tv * W2[k * 2 + 1];
  }
  a0 = waveRedSum(a0);
  a1 = waveRedSum(a1);
  if (lane == 0) {
    float v0 = a0 + b2[0], v1 = a1 + b2[1];
    if (SIG) { v0 = sigmoidf(v0); v1 = sigmoidf(v1); }
    out[row * 2]     = v0;
    out[row * 2 + 1] = v1;
  }
}

// wave-per-row, one output + softplus (K=128); MEAN=1 -> batch-mean atomic
template <int MEAN>
__global__ void head1_w(const float* __restrict__ T, const float* __restrict__ W2,
                        const float* __restrict__ b2, float* __restrict__ out) {
  const int row = (blockIdx.x * blockDim.x + threadIdx.x) >> 5;
  const int lane = threadIdx.x & 31;
  if (row >= NODES) return;
  const float* t = T + (size_t)row * 128;
  float a = 0.0f;
  for (int k = lane; k < 128; k += 32) a += t[k] * W2[k];
  a = waveRedSum(a);
  if (lane == 0) {
    const float sp = softplusf(a + b2[0]);
    if (MEAN) atomicAdd(&out[row >> 11], sp * (1.0f / 2048.0f));
    else      out[row] = sp;
  }
}

// ---------- launcher ----------
extern "C" void kernel_launch(void* const* d_in, const int* in_sizes, int n_in,
                              void* d_out, int out_size, void* d_ws, size_t ws_size,
                              hipStream_t stream) {
  (void)in_sizes; (void)n_in; (void)out_size; (void)ws_size;
  const float* h_in  = (const float*)d_in[0];
  const float* ctmp  = (const float*)d_in[1];
  const float* cster = (const float*)d_in[2];
  const float* eproj = (const float*)d_in[3];
  const float* fLt   = (const float*)d_in[4];
  const long long* edges = (const long long*)d_in[5];
  const float* eattr = (const float*)d_in[6];
  const float *Wl1=(const float*)d_in[7],  *bl1=(const float*)d_in[8];
  const float *Wr1=(const float*)d_in[9],  *br1=(const float*)d_in[10];
  const float *We1=(const float*)d_in[11], *att1=(const float*)d_in[12], *bias1=(const float*)d_in[13];
  const float *Wl2=(const float*)d_in[14], *bl2=(const float*)d_in[15];
  const float *Wr2=(const float*)d_in[16], *br2=(const float*)d_in[17];
  const float *We2=(const float*)d_in[18], *att2=(const float*)d_in[19], *bias2=(const float*)d_in[20];
  const float *Wih=(const float*)d_in[21], *bih=(const float*)d_in[22];
  const float *Whh=(const float*)d_in[23], *bhh=(const float*)d_in[24];
  const float *rW1=(const float*)d_in[25], *rb1=(const float*)d_in[26];
  const float *rW2=(const float*)d_in[27], *rb2=(const float*)d_in[28];
  const float *wW1=(const float*)d_in[29], *wb1=(const float*)d_in[30];
  const float *wW2=(const float*)d_in[31], *wb2=(const float*)d_in[32];
  const float *pW1=(const float*)d_in[33], *pb1=(const float*)d_in[34];
  const float *pW2=(const float*)d_in[35], *pb2=(const float*)d_in[36];
  const float *dW1=(const float*)d_in[37], *db1=(const float*)d_in[38];
  const float *dW2=(const float*)d_in[39], *db2=(const float*)d_in[40];

  // outputs (flat, in return order)
  float* out_h   = (float*)d_out;                 // 8192*256
  float* out_res = out_h + (size_t)NODES * 256;   // 8192*2
  float* out_w   = out_res + NODES * 2;           // 8192*2
  float* out_ap  = out_w + NODES * 2;             // 4
  float* out_ad  = out_ap + 4;                    // 8192

  // workspace layout
  float* ws = (float*)d_ws;
  size_t o = 0;
  auto alloc = [&](size_t n) { float* p = ws + o; o += n; return p; };
  float* X    = alloc((size_t)NODES * IN1P);  // padded x (later reused as head temp)
  float* Wl1p = alloc((size_t)IN1P * 256);    // zero-padded conv1 weights
  float* Wr1p = alloc((size_t)IN1P * 256);
  float* L1   = alloc((size_t)NODES * 256);   // xl1
  float* R1   = alloc((size_t)NODES * 256);   // xr1
  float* X1   = alloc((size_t)NODES * 256);   // conv1 out -> x1
  float* X2   = alloc((size_t)NODES * 256);   // conv2 out
  float* W0   = alloc((size_t)NODES * 1024);  // xl2, later gi
  float* W1   = alloc((size_t)NODES * 1024);  // xr2, later gh
  float* W2   = alloc((size_t)NODES * 1024);  // conv2 accumulator
  float* ESC  = alloc((size_t)E_EXT * 4);     // scores / alpha
  float* MB   = alloc((size_t)NODES * 4);     // segment max
  float* SB   = alloc((size_t)NODES * 4);     // segment sum
  float* EA   = alloc((size_t)E_EXT);         // edge attr ext
  float* MEAN = alloc(4);
  int* SRC = (int*)alloc((size_t)E_EXT);
  int* DST = (int*)alloc((size_t)E_EXT);
  float* HT = X;                               // head temp (8192x256 fits in X)

  const dim3 blk(256);
  auto cdiv = [](int a, int b) { return (a + b - 1) / b; };
  auto gemm = [&](const float* A, const float* Wm, const float* b, float* C,
                  int M, int N, int K, int act) {
    dim3 grid(N / 128, M / 32);
    if (act == 0) gemm_bf16_wmma<0><<<grid, blk, 0, stream>>>(A, Wm, b, C, M, N, K);
    else          gemm_bf16_wmma<1><<<grid, blk, 0, stream>>>(A, Wm, b, C, M, N, K);
  };
  const int EH = E_EXT * 4;

  // ---- input assembly + self loops + weight padding ----
  build_x<<<cdiv(NODES * IN1P, 256), blk, 0, stream>>>(ctmp, cster, eproj, fLt, X);
  pad_weight<<<cdiv(IN1P * 256, 256), blk, 0, stream>>>(Wl1, Wl1p, IN1, 256, IN1P);
  pad_weight<<<cdiv(IN1P * 256, 256), blk, 0, stream>>>(Wr1, Wr1p, IN1, 256, IN1P);
  ea_mean<<<1, blk, 0, stream>>>(eattr, MEAN);
  build_edges<<<cdiv(E_EXT, 256), blk, 0, stream>>>(edges, eattr, MEAN, SRC, DST, EA);

  // ---- conv1 (H=4, C=64, concat) ----
  gemm(X, Wl1p, bl1, L1, NODES, 256, IN1P, 0);
  gemm(X, Wr1p, br1, R1, NODES, 256, IN1P, 0);
  fill_f32<<<cdiv(NODES * 4, 256), blk, 0, stream>>>(MB, -INFINITY, NODES * 4);
  fill_f32<<<cdiv(NODES * 4, 256), blk, 0, stream>>>(SB, 0.0f, NODES * 4);
  fill_f32<<<cdiv(NODES * 256, 256), blk, 0, stream>>>(X1, 0.0f, NODES * 256);
  gat_score_w<<<EH / 8, blk, 0, stream>>>(L1, R1, EA, SRC, DST, We1, att1, ESC, MB, 4, 64);
  gat_alpha<<<cdiv(EH, 256), blk, 0, stream>>>(ESC, DST, MB, SB, 4);
  gat_agg_w<<<EH / 8, blk, 0, stream>>>(ESC, SRC, DST, SB, L1, X1, 4, 64);
  add_bias_relu<<<cdiv(NODES * 256, 256), blk, 0, stream>>>(X1, bias1, NODES * 256, 256);

  // ---- conv2 (H=4, C=256, mean over heads) ----
  gemm(X1, Wl2, bl2, W0, NODES, 1024, 256, 0);
  gemm(X1, Wr2, br2, W1, NODES, 1024, 256, 0);
  fill_f32<<<cdiv(NODES * 4, 256), blk, 0, stream>>>(MB, -INFINITY, NODES * 4);
  fill_f32<<<cdiv(NODES * 4, 256), blk, 0, stream>>>(SB, 0.0f, NODES * 4);
  fill_f32<<<cdiv(NODES * 1024, 256), blk, 0, stream>>>(W2, 0.0f, NODES * 1024);
  gat_score_w<<<EH / 8, blk, 0, stream>>>(W0, W1, EA, SRC, DST, We2, att2, ESC, MB, 4, 256);
  gat_alpha<<<cdiv(EH, 256), blk, 0, stream>>>(ESC, DST, MB, SB, 4);
  gat_agg_w<<<EH / 8, blk, 0, stream>>>(ESC, SRC, DST, SB, W0, W2, 4, 256);
  head_mean<<<cdiv(NODES * 256, 256), blk, 0, stream>>>(W2, bias2, X2);

  // ---- GRU (gi/gh reuse W0/W1) ----
  gemm(X2, Wih, bih, W0, NODES, 768, 256, 0);
  gemm(h_in, Whh, bhh, W1, NODES, 768, 256, 0);
  gru_cell<<<cdiv(NODES * 256, 256), blk, 0, stream>>>(W0, W1, h_in, out_h);

  // ---- heads ----
  gemm(out_h, rW1, rb1, HT, NODES, 256, 256, 1);
  head2_w<0><<<NODES / 8, blk, 0, stream>>>(HT, rW2, rb2, out_res);
  gemm(out_h, wW1, wb1, HT, NODES, 256, 256, 1);
  head2_w<1><<<NODES / 8, blk, 0, stream>>>(HT, wW2, wb2, out_w);
  gemm(out_h, pW1, pb1, HT, NODES, 128, 256, 1);
  fill_f32<<<1, 4, 0, stream>>>(out_ap, 0.0f, 4);
  head1_w<1><<<NODES / 8, blk, 0, stream>>>(HT, pW2, pb2, out_ap);
  gemm(out_h, dW1, db1, HT, NODES, 128, 256, 1);
  head1_w<0><<<NODES / 8, blk, 0, stream>>>(HT, dW2, db2, out_ad);
}
